// VAE_Model_51857435132201
// MI455X (gfx1250) — matrix operand order repaired and checked by reading, not compiled, not run on hardware
//
#include <hip/hip_runtime.h>
#include <hip/hip_bf16.h>
#include <math.h>

// ---------------------------------------------------------------------------
// Types for gfx1250 WMMA (wave32, v_wmma_f32_16x16x32_bf16)
// ---------------------------------------------------------------------------
typedef __bf16 bf16_t;
typedef bf16_t v16bf __attribute__((ext_vector_type(16)));
typedef bf16_t v8bf  __attribute__((ext_vector_type(8)));
typedef float  v8f   __attribute__((ext_vector_type(8)));
typedef int    v4i   __attribute__((ext_vector_type(4)));

// Model constants
#define BATCH 512
#define ALPH  40
#define SEQL  112
#define NCKC  64
#define FLATK 5632   // 88*64, multiple of 64
#define FNNN  512
#define LATN  292
#define LATP  320    // ceil32(292), also multiple of 64
#define GHN   512
#define G3N   1536
#define PRN   256
#define TSTEP 111

#define BS_STRIDE 72  // 64 K-elems + 8 pad (144B rows -> conflict-free-ish banks)

#if defined(__has_builtin)
#if __has_builtin(__builtin_amdgcn_global_load_async_to_lds_b128)
#define HAVE_ASYNC_LDS 1
#endif
#endif

#if defined(HAVE_ASYNC_LDS)
// clang prints AS1 as `__device__` and AS3 as `__shared__` in HIP mode; the
// builtin takes (int4 AS1* global_src, int4 AS3* lds_dst, int off, int cpol).
typedef __attribute__((address_space(1))) v4i* as1_v4i_p;
typedef __attribute__((address_space(3))) v4i* as3_v4i_p;
#endif

// ---------------------------------------------------------------------------
// WMMA GEMM body: C[M,N] = A[M,Kp](bf16) @ W[Np,Kp]^T + bias
// 256 threads = 8 waves; wave w -> 16-row strip; 4 accumulators -> 64 cols.
// Weight tile (64 N-rows x 64 K) staged in LDS once per block per K-chunk
// (async copy to LDS when available). Kp must be a multiple of 64.
// Grid: (ceil(N/64), M/128 [, problems]). Np padded to 64 with zeros.
// ---------------------------------------------------------------------------
__device__ __forceinline__
void gemm_body(const bf16_t* __restrict__ A, const bf16_t* __restrict__ W,
               const float* __restrict__ bias, float* __restrict__ C,
               int M, int N, int Kp, int ldc, bf16_t* Bs)
{
    const int lane = threadIdx.x & 31;
    const int wave = threadIdx.x >> 5;   // 0..7
    const int h    = lane >> 4;          // half: 0/1
    const int r    = lane & 15;
    const int mBase = blockIdx.y * 128 + wave * 16;
    const int nBase = blockIdx.x * 64;

    v8f acc[4];
#pragma unroll
    for (int nt = 0; nt < 4; ++nt)
#pragma unroll
        for (int i = 0; i < 8; ++i) acc[nt][i] = 0.0f;

    const bf16_t* aRow = A + (size_t)(mBase + r) * Kp;

    for (int k0 = 0; k0 < Kp; k0 += 64) {
        // ---- stage weight tile: 64 rows x 64 K bf16 = 8KB (512 x 16B chunks)
#pragma unroll
        for (int s = threadIdx.x; s < 512; s += 256) {
            int brow = s >> 3, bseg = s & 7;          // 8 x 16B per row
            const bf16_t* gp = W + (size_t)(nBase + brow) * Kp + k0 + bseg * 8;
            bf16_t* lp = Bs + brow * BS_STRIDE + bseg * 8;
#if defined(HAVE_ASYNC_LDS)
            __builtin_amdgcn_global_load_async_to_lds_b128(
                (as1_v4i_p)gp, (as3_v4i_p)lp, 0, 0);
#else
            *(v8bf*)lp = *(const v8bf*)gp;
#endif
        }
#if defined(HAVE_ASYNC_LDS)
#if __has_builtin(__builtin_amdgcn_s_wait_asynccnt)
        __builtin_amdgcn_s_wait_asynccnt(0);
#else
        asm volatile("s_wait_asynccnt 0x0" ::: "memory");
#endif
#endif
        __syncthreads();

#pragma unroll
        for (int kk = 0; kk < 64; kk += 32) {
            // A fragment: 16x32 bf16 per ISA layout (two 16B chunks per lane)
            v8bf alo = *(const v8bf*)(aRow + k0 + kk + 8 * h);
            v8bf ahi = *(const v8bf*)(aRow + k0 + kk + 16 + 8 * h);
            v16bf af;
#pragma unroll
            for (int e = 0; e < 8; ++e) { af[e] = alo[e]; af[e + 8] = ahi[e]; }

#pragma unroll
            for (int nt = 0; nt < 4; ++nt) {
                // B fragment from LDS: lane=col(row of W), K = e + 8*h (e<8), 16 + 8*h + (e-8) (e>=8)
                const bf16_t* lrow = Bs + (nt * 16 + r) * BS_STRIDE + kk + 8 * h;
                v8bf blo = *(const v8bf*)lrow;
                v8bf bhi = *(const v8bf*)(lrow + 16);
                v16bf bfr;
#pragma unroll
                for (int e = 0; e < 8; ++e) { bfr[e] = blo[e]; bfr[e + 8] = bhi[e]; }
                acc[nt] = __builtin_amdgcn_wmma_f32_16x16x32_bf16(
                    false, af, false, bfr, (short)0, acc[nt], false, false);
            }
        }
        __syncthreads();
    }

#pragma unroll
    for (int nt = 0; nt < 4; ++nt) {
        int col = nBase + nt * 16 + r;
        if (col < N) {
            float bv = bias ? bias[col] : 0.0f;
#pragma unroll
            for (int i = 0; i < 8; ++i) {
                int row = mBase + i + 8 * h;      // C layout: VGPR i -> M = i + 8*half
                if (row < M) C[(size_t)row * ldc + col] = acc[nt][i] + bv;
            }
        }
    }
}

__global__ __launch_bounds__(256)
void gemm_wmma_bf16(const bf16_t* __restrict__ A, const bf16_t* __restrict__ W,
                    const float* __restrict__ bias, float* __restrict__ C,
                    int M, int N, int Kp, int ldc)
{
    __shared__ bf16_t Bs[64 * BS_STRIDE];
    gemm_body(A, W, bias, C, M, N, Kp, ldc, Bs);
}

// Two independent GEMMs (same M,N,Kp,ldc) in one launch; blockIdx.z selects.
__global__ __launch_bounds__(256)
void gemm_wmma_bf16_dual(const bf16_t* __restrict__ A0, const bf16_t* __restrict__ W0,
                         const float* __restrict__ b0, float* __restrict__ C0,
                         const bf16_t* __restrict__ A1, const bf16_t* __restrict__ W1,
                         const float* __restrict__ b1, float* __restrict__ C1,
                         int M, int N, int Kp, int ldc)
{
    __shared__ bf16_t Bs[64 * BS_STRIDE];
    if (blockIdx.z == 0) gemm_body(A0, W0, b0, C0, M, N, Kp, ldc, Bs);
    else                 gemm_body(A1, W1, b1, C1, M, N, Kp, ldc, Bs);
}

// ---------------------------------------------------------------------------
// f32 -> bf16 convert with zero padding: src (N,K) -> dst (Np,Kp)
// ---------------------------------------------------------------------------
__global__ void cvt_pad_bf16(const float* __restrict__ src, bf16_t* __restrict__ dst,
                             int N, int K, int Np, int Kp)
{
    int idx = blockIdx.x * 256 + threadIdx.x;
    int total = Np * Kp;
    if (idx >= total) return;
    int nr = idx / Kp, kc = idx % Kp;
    float v = (nr < N && kc < K) ? src[(size_t)nr * K + kc] : 0.0f;
    dst[idx] = (bf16_t)v;
}

// ---------------------------------------------------------------------------
// 1D VALID conv, NCH layout, kernel width 9, optional leaky relu
// ---------------------------------------------------------------------------
__global__ void conv1d_k9(const float* __restrict__ x, const float* __restrict__ w,
                          const float* __restrict__ b, float* __restrict__ y,
                          int Cin, int Lin, int OC, int Lout, int doLrelu)
{
    int idx = blockIdx.x * 256 + threadIdx.x;
    int total = BATCH * OC * Lout;
    if (idx >= total) return;
    int t  = idx % Lout;
    int oc = (idx / Lout) % OC;
    int bb = idx / (Lout * OC);
    const float* xp = x + (size_t)bb * Cin * Lin;
    const float* wp = w + (size_t)oc * Cin * 9;
    float s = b[oc];
    for (int c = 0; c < Cin; ++c) {
        const float* xc = xp + c * Lin + t;
        const float* wc = wp + c * 9;
#pragma unroll
        for (int k = 0; k < 9; ++k) s += xc[k] * wc[k];
    }
    if (doLrelu) s = (s >= 0.0f) ? s : 0.2f * s;
    y[idx] = s;
}

// per-channel BN stats over (batch, spatial)
__global__ void bnc_stats(const float* __restrict__ y, float* mean, float* var,
                          int C, int L)
{
    int c = blockIdx.x;
    __shared__ float ss[256], sq[256];
    float s = 0.f, q = 0.f;
    int Bt = BATCH * L;
    for (int i = threadIdx.x; i < Bt; i += 256) {
        int bb = i / L, t = i % L;
        float v = y[((size_t)bb * C + c) * L + t];
        s += v; q += v * v;
    }
    ss[threadIdx.x] = s; sq[threadIdx.x] = q;
    __syncthreads();
    for (int st = 128; st > 0; st >>= 1) {
        if ((int)threadIdx.x < st) { ss[threadIdx.x] += ss[threadIdx.x + st]; sq[threadIdx.x] += sq[threadIdx.x + st]; }
        __syncthreads();
    }
    if (threadIdx.x == 0) {
        float m = ss[0] / (float)Bt;
        mean[c] = m;
        var[c]  = sq[0] / (float)Bt - m * m;
    }
}

__global__ void bnc_apply(float* __restrict__ y, const float* mean, const float* var,
                          const float* g, const float* beta, int C, int L, int doLrelu)
{
    int idx = blockIdx.x * 256 + threadIdx.x;
    int total = BATCH * C * L;
    if (idx >= total) return;
    int c = (idx / L) % C;
    float v = g[c] * (y[idx] - mean[c]) * rsqrtf(var[c] + 1e-5f) + beta[c];
    if (doLrelu) v = (v >= 0.0f) ? v : 0.2f * v;
    y[idx] = v;
}

// per-feature BN stats over batch dim (x is (BATCH, N) row-major)
__global__ void bnf_stats(const float* __restrict__ x, float* mean, float* var, int N)
{
    int c = blockIdx.x;
    __shared__ float ss[256], sq[256];
    float s = 0.f, q = 0.f;
    for (int i = threadIdx.x; i < BATCH; i += 256) {
        float v = x[(size_t)i * N + c];
        s += v; q += v * v;
    }
    ss[threadIdx.x] = s; sq[threadIdx.x] = q;
    __syncthreads();
    for (int st = 128; st > 0; st >>= 1) {
        if ((int)threadIdx.x < st) { ss[threadIdx.x] += ss[threadIdx.x + st]; sq[threadIdx.x] += sq[threadIdx.x + st]; }
        __syncthreads();
    }
    if (threadIdx.x == 0) {
        float m = ss[0] / (float)BATCH;
        mean[c] = m;
        var[c]  = sq[0] / (float)BATCH - m * m;
    }
}

// BN apply + leaky relu, emit bf16 activations padded to Kp
__global__ void bnf_apply_bf16(const float* __restrict__ x, const float* mean,
                               const float* var, const float* g, const float* beta,
                               bf16_t* __restrict__ out, int N, int Kp)
{
    int idx = blockIdx.x * 256 + threadIdx.x;
    int total = BATCH * Kp;
    if (idx >= total) return;
    int bb = idx / Kp, c = idx % Kp;
    if (c < N) {
        float v = g[c] * (x[(size_t)bb * N + c] - mean[c]) * rsqrtf(var[c] + 1e-5f) + beta[c];
        v = (v >= 0.0f) ? v : 0.2f * v;
        out[idx] = (bf16_t)v;
    } else out[idx] = (bf16_t)0.0f;
}

// std = softplus(raw)+1e-6 ; z = std*eps + mu ; bf16 copies of z and mu (padded)
__global__ void mu_std_z(const float* __restrict__ mu, const float* __restrict__ stdRaw,
                         const float* __restrict__ eps, float* __restrict__ stdOut,
                         bf16_t* __restrict__ zBf, bf16_t* __restrict__ muBf)
{
    int idx = blockIdx.x * 256 + threadIdx.x;
    int total = BATCH * LATP;
    if (idx >= total) return;
    int bb = idx / LATP, c = idx % LATP;
    if (c < LATN) {
        size_t src = (size_t)bb * LATN + c;
        float m  = mu[src];
        float sr = stdRaw[src];
        float sp = fmaxf(sr, 0.0f) + log1pf(expf(-fabsf(sr))) + 1e-6f;
        stdOut[src] = sp;
        float z = sp * eps[src] + m;
        zBf[idx]  = (bf16_t)z;
        muBf[idx] = (bf16_t)m;
    } else { zBf[idx] = (bf16_t)0.0f; muBf[idx] = (bf16_t)0.0f; }
}

// broadcast h0 to 3 GRU layers (f32 state + bf16 copy)
__global__ void init_hidden(const float* __restrict__ h0,
                            float* hf0, float* hf1, float* hf2,
                            bf16_t* hb0, bf16_t* hb1, bf16_t* hb2)
{
    int idx = blockIdx.x * 256 + threadIdx.x;
    if (idx >= BATCH * GHN) return;
    float v = h0[idx];
    bf16_t vb = (bf16_t)v;
    hf0[idx] = v; hf1[idx] = v; hf2[idx] = v;
    hb0[idx] = vb; hb1[idx] = vb; hb2[idx] = vb;
}

// tokens[b] = argmax_a model_input[b, a, 0]
__global__ void init_tokens(const float* __restrict__ x, int* __restrict__ tok)
{
    int b = blockIdx.x * 256 + threadIdx.x;
    if (b >= BATCH) return;
    const float* xp = x + (size_t)b * ALPH * SEQL;
    int best = 0; float bv = xp[0];
    for (int a = 1; a < ALPH; ++a) {
        float v = xp[(size_t)a * SEQL];
        if (v > bv) { bv = v; best = a; }
    }
    tok[b] = best;
}

// Layer-0 GRU gate, with one-hot input matmul fused as a column gather:
// gi0[b,n] = wi0[n, tok[b]] + bi0[n]
__global__ void gru_gate0_fused(const float* __restrict__ wi0, const float* __restrict__ bi0,
                                const int* __restrict__ tok, const float* __restrict__ gh,
                                float* __restrict__ hf, bf16_t* __restrict__ hb)
{
    int idx = blockIdx.x * 256 + threadIdx.x;
    if (idx >= BATCH * GHN) return;
    int b = idx >> 9, j = idx & 511;
    int tk = tok[b];
    const float* ghb = gh + (size_t)b * G3N;
    float gir = wi0[(size_t)j * ALPH + tk] + bi0[j];
    float giz = wi0[(size_t)(512 + j) * ALPH + tk] + bi0[512 + j];
    float gin = wi0[(size_t)(1024 + j) * ALPH + tk] + bi0[1024 + j];
    float r = 1.0f / (1.0f + expf(-(gir + ghb[j])));
    float z = 1.0f / (1.0f + expf(-(giz + ghb[512 + j])));
    float n = tanhf(gin + r * ghb[1024 + j]);
    float ho = hf[idx];
    float hn = (1.0f - z) * n + z * ho;
    hf[idx] = hn;
    hb[idx] = (bf16_t)hn;
}

// fused GRU gate update; writes f32 state + bf16 copy
__global__ void gru_gate(const float* __restrict__ gi, const float* __restrict__ gh,
                         float* __restrict__ hf, bf16_t* __restrict__ hb)
{
    int idx = blockIdx.x * 256 + threadIdx.x;
    if (idx >= BATCH * GHN) return;
    int b = idx >> 9, j = idx & 511;
    const float* gib = gi + (size_t)b * G3N;
    const float* ghb = gh + (size_t)b * G3N;
    float r = 1.0f / (1.0f + expf(-(gib[j] + ghb[j])));
    float z = 1.0f / (1.0f + expf(-(gib[512 + j] + ghb[512 + j])));
    float n = tanhf(gib[1024 + j] + r * ghb[1024 + j]);
    float ho = hf[idx];
    float hn = (1.0f - z) * n + z * ho;
    hf[idx] = hn;
    hb[idx] = (bf16_t)hn;
}

// argmax feedback + scatter logits into gru_output (B, 40, 111) at step t
__global__ void post_argmax_write(const float* __restrict__ logits, int* __restrict__ tok,
                                  float* __restrict__ outGru, int t)
{
    int b = blockIdx.x;
    if (threadIdx.x == 0) {
        const float* lp = logits + (size_t)b * ALPH;
        int best = 0; float bv = lp[0];
        for (int a = 1; a < ALPH; ++a) { float v = lp[a]; if (v > bv) { bv = v; best = a; } }
        tok[b] = best;
    }
    if ((int)threadIdx.x < ALPH) {
        outGru[(size_t)b * ALPH * TSTEP + (size_t)threadIdx.x * TSTEP + t] =
            logits[(size_t)b * ALPH + threadIdx.x];
    }
}

// ---------------------------------------------------------------------------
// Host orchestration
// ---------------------------------------------------------------------------
enum {
    IN_X = 0, IN_EPS = 1,
    C0_B = 2, C0_BETA = 3, C0_G = 4, C0_W = 5,
    C1_B = 6, C1_BETA = 7, C1_G = 8, C1_W = 9,
    C2_B = 10, C2_BETA = 11, C2_G = 12, C2_W = 13,
    DEC1_B = 14, DEC1_BETA = 15, DEC1_G = 16, DEC1_W = 17,
    DECO_B = 18, DECO_W = 19,
    E0_B = 20, E0_BETA = 21, E0_G = 22, E0_W = 23,
    E1_B = 24, E1_BETA = 25, E1_G = 26, E1_W = 27,
    G0_BH = 28, G0_BI = 29, G0_WH = 30, G0_WI = 31,
    G1_BH = 32, G1_BI = 33, G1_WH = 34, G1_WI = 35,
    G2_BH = 36, G2_BI = 37, G2_WH = 38, G2_WI = 39,
    MU_B = 40, MU_W = 41,
    POST_B = 42, POST_W = 43,
    PF0_B = 44, PF0_BETA = 45, PF0_G = 46, PF0_W = 47,
    PF1_B = 48, PF1_BETA = 49, PF1_G = 50, PF1_W = 51,
    POUT_B = 52, POUT_W = 53,
    STD_B = 54, STD_W = 55
};

static inline int ceilDiv(int a, int b) { return (a + b - 1) / b; }

extern "C" void kernel_launch(void* const* d_in, const int* in_sizes, int n_in,
                              void* d_out, int out_size, void* d_ws, size_t ws_size,
                              hipStream_t stream)
{
    (void)in_sizes; (void)out_size; (void)ws_size;
    char* wsp = (char*)d_ws;
    size_t off = 0;
    auto alloc = [&](size_t bytes) -> void* {
        void* p = wsp + off;
        off = (off + bytes + 255) & ~(size_t)255;
        return p;
    };

    // ---- bf16 padded weight buffers (converted each call; deterministic) ----
    bf16_t* wEnc1 = (bf16_t*)alloc((size_t)512 * FLATK * 2);
    bf16_t* wEnc2 = (bf16_t*)alloc((size_t)512 * 512 * 2);
    bf16_t* wMu   = (bf16_t*)alloc((size_t)320 * 512 * 2);
    bf16_t* wStd  = (bf16_t*)alloc((size_t)320 * 512 * 2);
    bf16_t* wDec1 = (bf16_t*)alloc((size_t)512 * LATP * 2);
    bf16_t* wDecO = (bf16_t*)alloc((size_t)512 * 512 * 2);
    bf16_t* wWh[3], *wWi12[2];
    for (int l = 0; l < 3; ++l) wWh[l] = (bf16_t*)alloc((size_t)G3N * 512 * 2);
    for (int l = 0; l < 2; ++l) wWi12[l] = (bf16_t*)alloc((size_t)G3N * 512 * 2);
    bf16_t* wPost = (bf16_t*)alloc((size_t)64 * 512 * 2);
    bf16_t* wPf1  = (bf16_t*)alloc((size_t)256 * LATP * 2);
    bf16_t* wPf2  = (bf16_t*)alloc((size_t)256 * 256 * 2);
    bf16_t* wPOut = (bf16_t*)alloc((size_t)64 * 256 * 2);

    // ---- activation / scratch buffers ----
    float* bufA  = (float*)alloc((size_t)BATCH * NCKC * 104 * 4);
    float* bufB  = (float*)alloc((size_t)BATCH * NCKC * 104 * 4);
    bf16_t* actBf  = (bf16_t*)alloc((size_t)BATCH * FLATK * 2);
    float* fcRaw  = (float*)alloc((size_t)BATCH * G3N * 4);
    float* stdRaw = (float*)alloc((size_t)BATCH * LATN * 4);
    bf16_t* enc1Bf = (bf16_t*)alloc((size_t)BATCH * 512 * 2);
    bf16_t* enc2Bf = (bf16_t*)alloc((size_t)BATCH * 512 * 2);
    bf16_t* zBf    = (bf16_t*)alloc((size_t)BATCH * LATP * 2);
    bf16_t* muBf   = (bf16_t*)alloc((size_t)BATCH * LATP * 2);
    bf16_t* dec1Bf = (bf16_t*)alloc((size_t)BATCH * 512 * 2);
    bf16_t* pf1Bf  = (bf16_t*)alloc((size_t)BATCH * 256 * 2);
    bf16_t* pf2Bf  = (bf16_t*)alloc((size_t)BATCH * 256 * 2);
    float* giBuf  = (float*)alloc((size_t)BATCH * G3N * 4);
    float* ghBuf  = (float*)alloc((size_t)BATCH * G3N * 4);
    float* hidF[3]; bf16_t* hidB[3];
    for (int l = 0; l < 3; ++l) hidF[l] = (float*)alloc((size_t)BATCH * GHN * 4);
    for (int l = 0; l < 3; ++l) hidB[l] = (bf16_t*)alloc((size_t)BATCH * GHN * 2);
    float* logits = (float*)alloc((size_t)BATCH * 64 * 4);
    int*   tokens = (int*)alloc((size_t)BATCH * 4);
    float* meanB  = (float*)alloc((size_t)G3N * 4);
    float* varB   = (float*)alloc((size_t)G3N * 4);

    float* outMu   = (float*)d_out;                               // (512,292)
    float* outStd  = outMu + (size_t)BATCH * LATN;                // (512,292)
    float* outGru  = outStd + (size_t)BATCH * LATN;               // (512,40,111)
    float* outProp = outGru + (size_t)BATCH * ALPH * TSTEP;       // (512,3)

    const float* fin[64];
    for (int i = 0; i < n_in && i < 64; ++i) fin[i] = (const float*)d_in[i];

    auto cvt = [&](int inIdx, bf16_t* dst, int N, int K, int Np, int Kp) {
        int total = Np * Kp;
        cvt_pad_bf16<<<dim3(ceilDiv(total, 256)), 256, 0, stream>>>(fin[inIdx], dst, N, K, Np, Kp);
    };
    auto gemm = [&](const bf16_t* A_, const bf16_t* W_, const float* bias,
                    float* C_, int N, int Kp, int ldc) {
        dim3 g(ceilDiv(N, 64), BATCH / 128);
        gemm_wmma_bf16<<<g, 256, 0, stream>>>(A_, W_, bias, C_, BATCH, N, Kp, ldc);
    };
    auto gemm2 = [&](const bf16_t* A0, const bf16_t* W0, const float* b0, float* C0,
                     const bf16_t* A1, const bf16_t* W1, const float* b1, float* C1,
                     int N, int Kp, int ldc) {
        dim3 g(ceilDiv(N, 64), BATCH / 128, 2);
        gemm_wmma_bf16_dual<<<g, 256, 0, stream>>>(A0, W0, b0, C0, A1, W1, b1, C1,
                                                   BATCH, N, Kp, ldc);
    };
    auto ew = [&](int total) { return dim3(ceilDiv(total, 256)); };

    // ---- weight conversion ----
    cvt(E0_W,   wEnc1,  512, FLATK, 512, FLATK);
    cvt(E1_W,   wEnc2,  512, 512,   512, 512);
    cvt(MU_W,   wMu,    LATN, 512,  320, 512);
    cvt(STD_W,  wStd,   LATN, 512,  320, 512);
    cvt(DEC1_W, wDec1,  512, LATN,  512, LATP);
    cvt(DECO_W, wDecO,  512, 512,   512, 512);
    cvt(G0_WH,  wWh[0], G3N, 512,   G3N, 512);
    cvt(G1_WH,  wWh[1], G3N, 512,   G3N, 512);
    cvt(G2_WH,  wWh[2], G3N, 512,   G3N, 512);
    cvt(G1_WI,  wWi12[0], G3N, 512, G3N, 512);
    cvt(G2_WI,  wWi12[1], G3N, 512, G3N, 512);
    cvt(POST_W, wPost,  ALPH, 512,  64,  512);
    cvt(PF0_W,  wPf1,   PRN, LATN,  256, LATP);
    cvt(PF1_W,  wPf2,   PRN, PRN,   256, 256);
    cvt(POUT_W, wPOut,  3,   PRN,   64,  256);

    // ---- conv encoder ----
    conv1d_k9<<<ew(BATCH * 64 * 104), 256, 0, stream>>>(fin[IN_X], fin[C0_W], fin[C0_B], bufA, ALPH, 112, 64, 104, 1);
    bnc_stats<<<64, 256, 0, stream>>>(bufA, meanB, varB, 64, 104);
    bnc_apply<<<ew(BATCH * 64 * 104), 256, 0, stream>>>(bufA, meanB, varB, fin[C0_G], fin[C0_BETA], 64, 104, 0);
    conv1d_k9<<<ew(BATCH * 64 * 96), 256, 0, stream>>>(bufA, fin[C1_W], fin[C1_B], bufB, 64, 104, 64, 96, 0);
    bnc_stats<<<64, 256, 0, stream>>>(bufB, meanB, varB, 64, 96);
    bnc_apply<<<ew(BATCH * 64 * 96), 256, 0, stream>>>(bufB, meanB, varB, fin[C1_G], fin[C1_BETA], 64, 96, 1);
    conv1d_k9<<<ew(BATCH * 64 * 88), 256, 0, stream>>>(bufB, fin[C2_W], fin[C2_B], bufA, 64, 96, 64, 88, 0);
    bnc_stats<<<64, 256, 0, stream>>>(bufA, meanB, varB, 64, 88);
    bnc_apply<<<ew(BATCH * 64 * 88), 256, 0, stream>>>(bufA, meanB, varB, fin[C2_G], fin[C2_BETA], 64, 88, 1);
    cvt_pad_bf16<<<ew(BATCH * FLATK), 256, 0, stream>>>(bufA, actBf, BATCH, FLATK, BATCH, FLATK);

    // ---- encoder FC ----
    gemm(actBf, wEnc1, fin[E0_B], fcRaw, 512, FLATK, 512);
    bnf_stats<<<512, 256, 0, stream>>>(fcRaw, meanB, varB, 512);
    bnf_apply_bf16<<<ew(BATCH * 512), 256, 0, stream>>>(fcRaw, meanB, varB, fin[E0_G], fin[E0_BETA], enc1Bf, 512, 512);
    gemm(enc1Bf, wEnc2, fin[E1_B], fcRaw, 512, 512, 512);
    bnf_stats<<<512, 256, 0, stream>>>(fcRaw, meanB, varB, 512);
    bnf_apply_bf16<<<ew(BATCH * 512), 256, 0, stream>>>(fcRaw, meanB, varB, fin[E1_G], fin[E1_BETA], enc2Bf, 512, 512);

    // ---- mu / std / z (mu and std GEMMs are independent -> dual launch) ----
    gemm2(enc2Bf, wMu, fin[MU_B], outMu,
          enc2Bf, wStd, fin[STD_B], stdRaw, LATN, 512, LATN);
    mu_std_z<<<ew(BATCH * LATP), 256, 0, stream>>>(outMu, stdRaw, fin[IN_EPS], outStd, zBf, muBf);

    // ---- decoder -> GRU init ----
    gemm(zBf, wDec1, fin[DEC1_B], fcRaw, 512, LATP, 512);
    bnf_stats<<<512, 256, 0, stream>>>(fcRaw, meanB, varB, 512);
    bnf_apply_bf16<<<ew(BATCH * 512), 256, 0, stream>>>(fcRaw, meanB, varB, fin[DEC1_G], fin[DEC1_BETA], dec1Bf, 512, 512);
    gemm(dec1Bf, wDecO, fin[DECO_B], fcRaw, 512, 512, 512);
    init_hidden<<<ew(BATCH * GHN), 256, 0, stream>>>(fcRaw, hidF[0], hidF[1], hidF[2], hidB[0], hidB[1], hidB[2]);
    init_tokens<<<ew(BATCH), 256, 0, stream>>>(fin[IN_X], tokens);

    // ---- GRU: 111 sequential steps, 8 launches each ----
    const float* biL[3] = { fin[G0_BI], fin[G1_BI], fin[G2_BI] };
    const float* bhL[3] = { fin[G0_BH], fin[G1_BH], fin[G2_BH] };
    for (int t = 0; t < TSTEP; ++t) {
        // layer 0: gh via WMMA; one-hot gi fused into gate kernel
        gemm(hidB[0], wWh[0], bhL[0], ghBuf, G3N, 512, G3N);
        gru_gate0_fused<<<ew(BATCH * GHN), 256, 0, stream>>>(fin[G0_WI], biL[0], tokens, ghBuf, hidF[0], hidB[0]);
        // layer 1: gi (new hid0) and gh (old hid1) are independent -> dual
        gemm2(hidB[0], wWi12[0], biL[1], giBuf,
              hidB[1], wWh[1],  bhL[1], ghBuf, G3N, 512, G3N);
        gru_gate<<<ew(BATCH * GHN), 256, 0, stream>>>(giBuf, ghBuf, hidF[1], hidB[1]);
        // layer 2
        gemm2(hidB[1], wWi12[1], biL[2], giBuf,
              hidB[2], wWh[2],  bhL[2], ghBuf, G3N, 512, G3N);
        gru_gate<<<ew(BATCH * GHN), 256, 0, stream>>>(giBuf, ghBuf, hidF[2], hidB[2]);
        // logits + argmax feedback + scatter into gru_output
        gemm(hidB[2], wPost, fin[POST_B], logits, ALPH, 512, ALPH);
        post_argmax_write<<<BATCH, 64, 0, stream>>>(logits, tokens, outGru, t);
    }

    // ---- property regressor (from mu) ----
    gemm(muBf, wPf1, fin[PF0_B], fcRaw, PRN, LATP, PRN);
    bnf_stats<<<PRN, 256, 0, stream>>>(fcRaw, meanB, varB, PRN);
    bnf_apply_bf16<<<ew(BATCH * PRN), 256, 0, stream>>>(fcRaw, meanB, varB, fin[PF0_G], fin[PF0_BETA], pf1Bf, PRN, PRN);
    gemm(pf1Bf, wPf2, fin[PF1_B], fcRaw, PRN, PRN, PRN);
    bnf_stats<<<PRN, 256, 0, stream>>>(fcRaw, meanB, varB, PRN);
    bnf_apply_bf16<<<ew(BATCH * PRN), 256, 0, stream>>>(fcRaw, meanB, varB, fin[PF1_G], fin[PF1_BETA], pf2Bf, PRN, PRN);
    gemm(pf2Bf, wPOut, fin[POUT_B], outProp, 3, PRN, 3);
}
